// RecurrentConvLayer_Sparse_4105988735628
// MI455X (gfx1250) — compile-verified, matrix operand order checked by reading, not run on hardware
//
#include <hip/hip_runtime.h>
#include <hip/hip_bf16.h>

typedef __attribute__((ext_vector_type(16))) __bf16 v16bf;
typedef __attribute__((ext_vector_type(8)))  float  v8f;

// ---------------- constants (shapes fixed by setup_inputs) ----------------
#define BATCH   4
#define CIN     32
#define COUT    64
#define HIN     256
#define HOUT    128          // 256/2
#define HPREV   320
#define EY2     32           // ey/2
#define C1_K    800          // 32*25
#define C1_KC   25           // K chunks of 32
#define C1_KP   816          // padded LDS row (bf16 elems), 32B-aligned rows
#define C2_K    1152         // 128*9
#define C2_KC   36
#define C2_KP   1168
#define NPIX    (BATCH*HOUT*HOUT)            // 65536
#define HID_N   (BATCH*COUT*HOUT*HOUT)       // 4194304
#define PLANE_N (BATCH*COUT*HPREV*HPREV)     // 26214400

__device__ __forceinline__ unsigned short f32_to_bf16(float f) {
    unsigned int u = __float_as_uint(f);
    u += 0x7FFFu + ((u >> 16) & 1u);          // round-to-nearest-even
    return (unsigned short)(u >> 16);
}
__device__ __forceinline__ float sigm(float x) {
    return 1.0f / (1.0f + __expf(-x));
}
__device__ __forceinline__ float tanh_fast(float x) {
    return 2.0f / (1.0f + __expf(-2.0f * x)) - 1.0f;
}

// ---------------- weight packing: fragment-contiguous bf16 ----------------
// WcP[((nt*C1_KC + kc)*32 + lane)*16 + j] ; co = nt*16+(lane&15), k = kc*32+(lane>>4)*16+j
__global__ void pack_wc(const float* __restrict__ Wc, unsigned short* __restrict__ WcP) {
    int idx = blockIdx.x * 256 + threadIdx.x;
    if (idx >= 4 * C1_KC * 32 * 16) return;
    int j  = idx & 15;
    int l  = (idx >> 4) & 31;
    int kc = (idx >> 9) % C1_KC;
    int nt = idx / (16 * 32 * C1_KC);
    int co = nt * 16 + (l & 15);
    int k  = kc * 32 + (l >> 4) * 16 + j;
    int ci = k / 25, r = k - ci * 25, ky = r / 5, kx = r - ky * 5;
    WcP[idx] = f32_to_bf16(Wc[((co * CIN + ci) * 5 + ky) * 5 + kx]);
}

// WgP[((nt*C2_KC + kc)*32 + lane)*16 + j] ; cg = nt*16+(lane&15), k = kc*32+(lane>>4)*16+j
__global__ void pack_wg(const float* __restrict__ Wg, unsigned short* __restrict__ WgP) {
    int idx = blockIdx.x * 256 + threadIdx.x;
    if (idx >= 16 * C2_KC * 32 * 16) return;
    int j  = idx & 15;
    int l  = (idx >> 4) & 31;
    int kc = (idx >> 9) % C2_KC;
    int nt = idx / (16 * 32 * C2_KC);
    int cg = nt * 16 + (l & 15);
    int k  = kc * 32 + (l >> 4) * 16 + j;
    int cs = k / 9, r = k - cs * 9, ky = r / 3, kx = r - ky * 3;
    WgP[idx] = f32_to_bf16(Wg[((cg * 128 + cs) * 3 + ky) * 3 + kx]);
}

// ---------------- bulk copy prev_h/prev_c -> new_h/new_c ------------------
__global__ void copy_planes(const float4* __restrict__ ph, const float4* __restrict__ pc,
                            float4* __restrict__ nh, float4* __restrict__ nc, int n4) {
    int i = blockIdx.x * 256 + threadIdx.x;
    if (i < n4) { nh[i] = ph[i]; nc[i] = pc[i]; }
}

// ---------------- conv1: implicit GEMM, M=65536 N=64 K=800 ----------------
// block: 256 threads = 8 waves; 32 pixels x 64 channels per block.
// wave w: m-subtile (w>>2), n-tile (w&3). Output: ReLU(acc+bc) as bf16 -> ws.
__global__ void conv1_wmma(const float* __restrict__ x, const unsigned short* __restrict__ WcP,
                           const float* __restrict__ bc, unsigned short* __restrict__ xconv) {
    __shared__ __align__(32) unsigned short As[32 * C1_KP];   // 52224 B
    const int t = threadIdx.x;
    const int block_p = blockIdx.x * 32;

    // cooperative A-tile fill: stacked im2col rows (bf16), zero-pad border
    for (int idx = t; idx < 32 * C1_K; idx += 256) {
        int m = idx / C1_K, k = idx - m * C1_K;
        int p = block_p + m;
        int ox = p & 127, oy = (p >> 7) & 127, b = p >> 14;
        int ci = k / 25, r = k - ci * 25, ky = r / 5, kx = r - ky * 5;
        int iy = 2 * oy + ky - 2, ix = 2 * ox + kx - 2;
        float v = 0.0f;
        if ((unsigned)iy < (unsigned)HIN && (unsigned)ix < (unsigned)HIN)
            v = x[((b * CIN + ci) * HIN + iy) * HIN + ix];
        As[m * C1_KP + k] = f32_to_bf16(v);
    }
    __syncthreads();

    const int w = t >> 5, l = t & 31;
    const int mt = w >> 2, nt = w & 3;
    v8f acc = {};
    const unsigned short* abase = &As[(mt * 16 + (l & 15)) * C1_KP + ((l >> 4) * 16)];
    const unsigned short* bbase = WcP + (((nt * C1_KC) * 32 + l) * 16);
    for (int kc = 0; kc < C1_KC; ++kc) {
        v16bf a = *reinterpret_cast<const v16bf*>(abase + kc * 32);
        v16bf bfr = *reinterpret_cast<const v16bf*>(bbase + kc * (32 * 16));
        acc = __builtin_amdgcn_wmma_f32_16x16x32_bf16(false, a, false, bfr,
                                                      (short)0, acc, false, false);
    }

    // epilogue: +bias, ReLU, bf16, packed 16B store (8 consecutive ox per lane)
    const int co = nt * 16 + (l & 15);
    const float bias = bc[co];
    const int m0 = block_p + mt * 16 + 8 * (l >> 4);          // j = 0..7 consecutive
    const int ox0 = m0 & 127, oy = (m0 >> 7) & 127, b = m0 >> 14;
    unsigned int pk[4];
    #pragma unroll
    for (int d = 0; d < 4; ++d) {
        float v0 = acc[2 * d] + bias;     v0 = v0 > 0.0f ? v0 : 0.0f;
        float v1 = acc[2 * d + 1] + bias; v1 = v1 > 0.0f ? v1 : 0.0f;
        pk[d] = (unsigned int)f32_to_bf16(v0) | ((unsigned int)f32_to_bf16(v1) << 16);
    }
    uint4 st = {pk[0], pk[1], pk[2], pk[3]};
    *reinterpret_cast<uint4*>(&xconv[((b * COUT + co) * HOUT + oy) * HOUT + ox0]) = st;
}

// ------- conv2 fused with LSTM: M=65536 N=256 K=1152, gates stay in LDS ----
// block: 16 pixels x 256 gate channels; 8 waves, each owns n-tiles w and w+8.
__global__ void conv2_lstm_wmma(const unsigned short* __restrict__ xconv,
                                const float* __restrict__ prev_h,
                                const float* __restrict__ prev_c,
                                const unsigned short* __restrict__ WgP,
                                const float* __restrict__ bg,
                                float* __restrict__ out_hidden,
                                float* __restrict__ out_newh,
                                float* __restrict__ out_newc) {
    __shared__ __align__(32) unsigned short As[16 * C2_KP];   // 37376 B
    __shared__ float G[16 * 256];                             // 16384 B
    const int t = threadIdx.x;
    const int block_p = blockIdx.x * 16;

    // A tile: channels 0..63 from x_conv (zero-pad 1), 64..127 from prev_h crop
    for (int idx = t; idx < 16 * C2_K; idx += 256) {
        int m = idx / C2_K, k = idx - m * C2_K;
        int p = block_p + m;
        int ox = p & 127, oy = (p >> 7) & 127, b = p >> 14;
        int cs = k / 9, r = k - cs * 9, ky = r / 3, kx = r - ky * 3;
        unsigned short uv;
        if (cs < COUT) {
            int iy = oy + ky - 1, ix = ox + kx - 1;
            uv = 0;
            if ((unsigned)iy < (unsigned)HOUT && (unsigned)ix < (unsigned)HOUT)
                uv = xconv[((b * COUT + cs) * HOUT + iy) * HOUT + ix];
        } else {
            int hy = EY2 - 1 + oy + ky, hx = EY2 - 1 + ox + kx;   // always in-bounds
            uv = f32_to_bf16(prev_h[((b * COUT + (cs - COUT)) * HPREV + hy) * HPREV + hx]);
        }
        As[m * C2_KP + k] = uv;
    }
    __syncthreads();

    const int w = t >> 5, l = t & 31;
    v8f acc0 = {}, acc1 = {};
    const unsigned short* abase = &As[(l & 15) * C2_KP + ((l >> 4) * 16)];
    const unsigned short* b0 = WgP + (((w * C2_KC) * 32 + l) * 16);
    const unsigned short* b1 = WgP + ((((w + 8) * C2_KC) * 32 + l) * 16);
    for (int kc = 0; kc < C2_KC; ++kc) {
        v16bf a   = *reinterpret_cast<const v16bf*>(abase + kc * 32);
        v16bf bf0 = *reinterpret_cast<const v16bf*>(b0 + kc * (32 * 16));
        v16bf bf1 = *reinterpret_cast<const v16bf*>(b1 + kc * (32 * 16));
        acc0 = __builtin_amdgcn_wmma_f32_16x16x32_bf16(false, a, false, bf0,
                                                       (short)0, acc0, false, false);
        acc1 = __builtin_amdgcn_wmma_f32_16x16x32_bf16(false, a, false, bf1,
                                                       (short)0, acc1, false, false);
    }

    // gates (+bias) -> LDS
    {
        const int n = l & 15, mh = 8 * (l >> 4);
        const int c0 = w * 16 + n, c1 = (w + 8) * 16 + n;
        const float bia0 = bg[c0], bia1 = bg[c1];
        #pragma unroll
        for (int j = 0; j < 8; ++j) {
            G[(mh + j) * 256 + c0] = acc0[j] + bia0;
            G[(mh + j) * 256 + c1] = acc1[j] + bia1;
        }
    }
    __syncthreads();

    // pointwise LSTM: 16 pixels x 64 channels
    for (int idx = t; idx < 16 * COUT; idx += 256) {
        int m = idx >> 6, co = idx & 63;
        int p = block_p + m;
        int ox = p & 127, oy = (p >> 7) & 127, b = p >> 14;
        float ig = G[m * 256 + co];
        float rg = G[m * 256 + 64 + co];
        float og = G[m * 256 + 128 + co];
        float cg = G[m * 256 + 192 + co];
        long long ppos = ((long long)(b * COUT + co) * HPREV + (EY2 + oy)) * HPREV + (EY2 + ox);
        float pc   = prev_c[ppos];
        float cell = sigm(rg) * pc + sigm(ig) * tanh_fast(cg);
        float hid  = sigm(og) * tanh_fast(cell);
        out_hidden[((b * COUT + co) * HOUT + oy) * HOUT + ox] = hid;
        out_newh[ppos] = hid;
        out_newc[ppos] = cell;
    }
}

// ------------------------------- launch ------------------------------------
extern "C" void kernel_launch(void* const* d_in, const int* in_sizes, int n_in,
                              void* d_out, int out_size, void* d_ws, size_t ws_size,
                              hipStream_t stream) {
    const float* x      = (const float*)d_in[0];
    const float* prev_h = (const float*)d_in[1];
    const float* prev_c = (const float*)d_in[2];
    const float* Wc     = (const float*)d_in[3];
    const float* bc     = (const float*)d_in[4];
    const float* Wg     = (const float*)d_in[5];
    const float* bg     = (const float*)d_in[6];

    float* out    = (float*)d_out;
    float* hidden = out;                       // 4*64*128*128
    float* new_h  = out + HID_N;               // 4*64*320*320
    float* new_c  = new_h + PLANE_N;

    char* ws = (char*)d_ws;
    unsigned short* xconv = (unsigned short*)ws;                          // 16 MiB
    unsigned short* WcP   = (unsigned short*)(ws + (size_t)16777216);     // 100 KiB
    unsigned short* WgP   = (unsigned short*)(ws + (size_t)16777216 + 102400); // 576 KiB

    // 1) weight packs
    pack_wc<<<(4 * C1_KC * 32 * 16 + 255) / 256, 256, 0, stream>>>(Wc, WcP);
    pack_wg<<<(16 * C2_KC * 32 * 16 + 255) / 256, 256, 0, stream>>>(Wg, WgP);

    // 2) bulk plane copies (window overwritten later on same stream)
    const int n4 = PLANE_N / 4;
    copy_planes<<<(n4 + 255) / 256, 256, 0, stream>>>(
        (const float4*)prev_h, (const float4*)prev_c, (float4*)new_h, (float4*)new_c, n4);

    // 3) conv1 -> x_conv (bf16 in ws)
    conv1_wmma<<<NPIX / 32, 256, 0, stream>>>(x, WcP, bc, xconv);

    // 4) conv2 + LSTM fused, writes hidden + windows of new_h/new_c
    conv2_lstm_wmma<<<NPIX / 16, 256, 0, stream>>>(
        xconv, prev_h, prev_c, WgP, bg, hidden, new_h, new_c);
}
